// DVDMixer_36112085025134
// MI455X (gfx1250) — matrix operand order repaired and checked by reading, not compiled, not run on hardware
//
#include <hip/hip_runtime.h>
#include <hip/hip_bf16.h>

// DVDMixer fused kernel for gfx1250 (MI455X).
// B = 4096 rows, 16 rows per block, 256 threads (8 wave32 waves) per block.
// Fast path: weights + hidden_states pre-swizzled once per launch into d_ws in
// fragment-major f16 layout (one contiguous 32B load per lane per WMMA operand).

#define N_AGENTS 16
#define RNN_H    64
#define N_HEADS  4
#define GAT_D    32
#define EMB      32
#define SDIM     128
#define ROWS_PER_BLK 16
#define NTHREADS 256

typedef __attribute__((ext_vector_type(16))) _Float16 v16h;
typedef __attribute__((ext_vector_type(8)))  float    v8f;

__device__ __forceinline__ v8f wmma16(v16h a, v16h b, v8f c) {
  return __builtin_amdgcn_wmma_f32_16x16x32_f16(
      false, a, false, b, (short)0, c, false, false);
}
__device__ __forceinline__ float elu1(float x)  { return x > 0.f ? x : (__expf(x) - 1.f); }
__device__ __forceinline__ float lrelu(float x) { return x > 0.f ? x : 0.2f * x; }

// LDS byte offset of a __shared__ object (generic -> addrspace(3) -> int).
__device__ __forceinline__ unsigned lds_addr(const void* p) {
  return (unsigned)(unsigned long long)(__attribute__((address_space(3))) const void*)p;
}

// ---------------------------------------------------------------------------
// Prologue: repack f32 weights / hidden_states into fragment-major f16.
// Fragment = 32 lanes x 16 halfs (1 KB). Element i of lane L uses this code's
// contiguous k order:  A: K = 32*cs + (L>=16 ? 16:0) + i,               M = L&15
//                  B: K = 32*cs + (L<16 ? i : 16+i),              N = L&15
// ---------------------------------------------------------------------------
__global__ __launch_bounds__(256) void swz_w1s(const float* __restrict__ W,
                                               _Float16* __restrict__ o) {
  int fi = blockIdx.x * 256 + threadIdx.x;          // 524288 total
  int i = fi & 15, lane = (fi >> 4) & 31, cs = (fi >> 9) & 3, nt = fi >> 11;
  int k = 32 * cs + ((lane < 16) ? i : 16 + i);
  int n = nt * 16 + (lane & 15);
  o[fi] = (_Float16) W[(size_t)k * 4096 + n];
}

__global__ __launch_bounds__(256) void swz_hs(const float* __restrict__ H,
                                              _Float16* __restrict__ o) {
  size_t fi = (size_t)blockIdx.x * 256 + threadIdx.x;   // 4194304 total
  int i = (int)(fi & 15);
  int lane = (int)((fi >> 4) & 31);
  int cs = (int)((fi >> 9) & 1);
  size_t r = fi >> 10;
  int k = 32 * cs + ((lane >= 16) ? 16 : 0) + i;
  int m = lane & 15;
  o[fi] = (_Float16) H[(r * 16 + m) * 64 + k];
}

__global__ __launch_bounds__(256) void swz_small(const float* __restrict__ Wg,
                                                 const float* __restrict__ b1W,
                                                 const float* __restrict__ wfW,
                                                 const float* __restrict__ V1W,
                                                 _Float16* __restrict__ oWg,
                                                 _Float16* __restrict__ oSm) {
  int fi = blockIdx.x * 256 + threadIdx.x;          // 20480 total
  if (fi < 8192) {                                  // W_gat: 8 tiles x 2 ksteps
    int i = fi & 15, lane = (fi >> 4) & 31, cs = (fi >> 9) & 1, tile = fi >> 10;
    int k = 32 * cs + ((lane < 16) ? i : 16 + i);
    oWg[fi] = (_Float16) Wg[k * 128 + tile * 16 + (lane & 15)];
  } else {                                          // b1/wf/V1: 6 tiles x 4 ksteps
    int f2 = fi - 8192;
    int i = f2 & 15, lane = (f2 >> 4) & 31, cs = (f2 >> 9) & 3, rest = f2 >> 11;
    int which = rest >> 1, t = rest & 1;
    const float* Wm = (which == 0) ? b1W : (which == 1) ? wfW : V1W;
    int k = 32 * cs + ((lane < 16) ? i : 16 + i);
    oSm[f2] = (_Float16) Wm[k * EMB + t * 16 + (lane & 15)];
  }
}

// ---------------------------------------------------------------------------
// Main fused kernel. UW = use pre-swizzled workspace fragments.
// ---------------------------------------------------------------------------
template <bool UW>
__global__ __launch_bounds__(NTHREADS)
void dvd_mixer_fused(
    const float* __restrict__ qs,    // (B,16)
    const float* __restrict__ st,    // (B,128)
    const float* __restrict__ hs,    // (B,16,64)
    const float* __restrict__ unc,   // (B)
    const float* __restrict__ Wg,    // (64,128)
    const float* __restrict__ attA,  // (4,64)
    const float* __restrict__ w1sW,  // (129,4096)
    const float* __restrict__ w1sB,  // (4096)
    const float* __restrict__ b1W, const float* __restrict__ b1B,
    const float* __restrict__ wfW, const float* __restrict__ wfB,
    const float* __restrict__ V1W, const float* __restrict__ V1B,
    const float* __restrict__ V2W, const float* __restrict__ V2B,
    const _Float16* __restrict__ wsHS,   // fragment-major hs
    const _Float16* __restrict__ wsW1,   // fragment-major w1s_W
    const _Float16* __restrict__ wsWg,   // fragment-major W_gat
    const _Float16* __restrict__ wsSm,   // fragment-major b1/wf/V1
    float* __restrict__ out)
{
  __shared__ __align__(16) float sSt[16 * SDIM];        // states tile f32 (8 KB)
  __shared__ _Float16 sHP[16 * 16 * GAT_D];             // per-head hp (16 KB)
  __shared__ _Float16 sG [16 * 16 * GAT_D];             // per-head g  (16 KB)
  __shared__ _Float16 sCst[8 * 16 * GAT_D];             // per-wave C  (8 KB)
  __shared__ float sUnc[16];
  __shared__ float sQ[16 * 16];
  __shared__ float sSrc[16 * 16], sDst[16 * 16];
  __shared__ float sB1[16 * EMB], sWf[16 * EMB], sV1r[16 * EMB], sHid[16 * EMB];
  __shared__ float sV[16];
  __shared__ float sPart[8 * 2 * 16 * 4];

  const int tid  = threadIdx.x;
  const int lane = tid & 31;
  const int w    = tid >> 5;
  const int l16  = lane & 15;
  const bool hi  = lane >= 16;
  const int row0 = blockIdx.x * ROWS_PER_BLK;

  // ---- states tile -> LDS via async copy (ASYNCcnt path) ----
  {
    const float* gsrc = st + (size_t)row0 * SDIM;
    for (int i = tid; i < (16 * SDIM) / 4; i += NTHREADS) {
      unsigned loff = lds_addr(&sSt[i * 4]);
      unsigned long long ga = (unsigned long long)(const void*)(gsrc + i * 4);
      asm volatile("global_load_async_to_lds_b128 %0, %1, off"
                   :: "v"(loff), "v"(ga) : "memory");
    }
    asm volatile("s_wait_asynccnt 0" ::: "memory");
  }
  if (tid < 16) sUnc[tid] = unc[row0 + tid];
  sQ[tid] = qs[(size_t)row0 * 16 + tid];
  __syncthreads();

  // A fragments of s (16x128, 4 k-steps), shared by all GEMMs on s.
  v16h aF[4];
#pragma unroll
  for (int cs = 0; cs < 4; ++cs) {
#pragma unroll
    for (int i = 0; i < 16; ++i) {
      int k = 32 * cs + (hi ? 16 : 0) + i;
      aF[cs][i] = (_Float16) sSt[l16 * SDIM + k];
    }
  }

  // A fragments of this wave's two hs rows (hoisted out of the head loop).
  v16h ha[2][2];
#pragma unroll
  for (int rr = 0; rr < 2; ++rr) {
    const int r = w * 2 + rr;
    if (UW) {
#pragma unroll
      for (int cs = 0; cs < 2; ++cs)
        ha[rr][cs] = *(const v16h*)(wsHS + ((size_t)(row0 + r) * 2 + cs) * 512 + lane * 16);
    } else {
      const float* hsrow = hs + ((size_t)(row0 + r)) * N_AGENTS * RNN_H;
#pragma unroll
      for (int cs = 0; cs < 2; ++cs)
#pragma unroll
        for (int i = 0; i < 16; ++i) {
          int k = 32 * cs + (hi ? 16 : 0) + i;
          ha[rr][cs][i] = (_Float16) hsrow[l16 * RNN_H + k];
        }
    }
  }

  // ---- hypernet small GEMMs: b1 / wf / V1 (waves 0..5) ----
  if (w < 6) {
    const int which = w >> 1, t = w & 1;
    const float* Bb = (which == 0) ? b1B : (which == 1) ? wfB : V1B;
    v8f c = {};
#pragma unroll
    for (int cs = 0; cs < 4; ++cs) {
      v16h bf;
      if (UW) {
        bf = *(const v16h*)(wsSm + (((which * 2 + t) * 4 + cs) * 32 + lane) * 16);
      } else {
        const float* Wm = (which == 0) ? b1W : (which == 1) ? wfW : V1W;
#pragma unroll
        for (int i = 0; i < 16; ++i) {
          int k = 32 * cs + (hi ? 16 + i : i);
          bf[i] = (_Float16) Wm[k * EMB + t * 16 + l16];
        }
      }
      c = wmma16(aF[cs], bf, c);
    }
    const int n = t * 16 + l16;
    const float bias = Bb[n];
#pragma unroll
    for (int j = 0; j < 8; ++j) {
      int m = j + (hi ? 8 : 0);
      float v = c[j] + bias;
      if (which == 0)      sB1[m * EMB + n] = v;
      else if (which == 1) sWf[m * EMB + n] = fabsf(v);
      else                 sV1r[m * EMB + n] = fmaxf(v, 0.f);
    }
  }
  __syncthreads();
  if (tid < 16) {
    float acc = V2B[0];
    for (int e = 0; e < EMB; ++e) acc += sV1r[tid * EMB + e] * V2W[e];
    sV[tid] = acc;
  }

  float w1acc[32];
#pragma unroll
  for (int i = 0; i < 32; ++i) w1acc[i] = 0.f;

  // ---- per-head pipeline ----
  for (int h = 0; h < N_HEADS; ++h) {
    __syncthreads();
    __builtin_prefetch(wsW1 ? (const void*)(wsW1 + (size_t)h * 64 * 4 * 512)
                            : (const void*)(w1sW + (size_t)h * 1024), 0, 1);

    // step 1: hp_r = hs_r(16x64) @ Wg[:, h*32 : h*32+32]
#pragma unroll
    for (int rr = 0; rr < 2; ++rr) {
      const int r = w * 2 + rr;
#pragma unroll
      for (int t = 0; t < 2; ++t) {
        v8f c = {};
#pragma unroll
        for (int cs = 0; cs < 2; ++cs) {
          v16h bf;
          if (UW) {
            bf = *(const v16h*)(wsWg + ((((h * 2 + t) * 2 + cs) * 32) + lane) * 16);
          } else {
#pragma unroll
            for (int i = 0; i < 16; ++i) {
              int k = 32 * cs + (hi ? 16 + i : i);
              bf[i] = (_Float16) Wg[k * (N_HEADS * GAT_D) + h * GAT_D + t * 16 + l16];
            }
          }
          c = wmma16(ha[rr][cs], bf, c);
        }
#pragma unroll
        for (int j = 0; j < 8; ++j) {
          int m = j + (hi ? 8 : 0);
          sHP[(r * 16 + m) * GAT_D + t * 16 + l16] = (_Float16) c[j];
        }
      }
    }
    __syncthreads();

    // step 2: src/dst projections
    {
      const int r = tid >> 4, i = tid & 15;
      float sa = 0.f, da = 0.f;
      for (int d = 0; d < GAT_D; ++d) {
        float v = (float) sHP[(r * 16 + i) * GAT_D + d];
        sa += v * attA[h * 64 + d];
        da += v * attA[h * 64 + GAT_D + d];
      }
      sSrc[r * 16 + i] = sa;
      sDst[r * 16 + i] = da;
    }
    __syncthreads();

    // step 3: leaky-relu + softmax + aggregate + ELU -> sG
    {
      const int r = tid >> 4, i = tid & 15;
      const float si = sSrc[r * 16 + i];
      float ev[16];
      float mx = -1e30f;
      for (int j = 0; j < 16; ++j) {
        float e = lrelu(si + sDst[r * 16 + j]);
        ev[j] = e;
        mx = fmaxf(mx, e);
      }
      float sm = 0.f;
      for (int j = 0; j < 16; ++j) { ev[j] = __expf(ev[j] - mx); sm += ev[j]; }
      const float inv = 1.f / sm;
      for (int d = 0; d < GAT_D; ++d) {
        float acc = 0.f;
        for (int j = 0; j < 16; ++j) acc += ev[j] * (float) sHP[(r * 16 + j) * GAT_D + d];
        sG[(r * 16 + i) * GAT_D + d] = (_Float16) elu1(acc * inv);
      }
    }
    __syncthreads();

    // step 4: main GEMM (K=128 WMMA + unc rank-1 in VALU) fused with |C.g^T|
    for (int ei = 0; ei < 4; ++ei) {
      const int e = w * 4 + ei;
#pragma unroll
      for (int t = 0; t < 2; ++t) {
        const int ntile = h * 64 + e * 2 + t;
        const int ncol  = ntile * 16 + l16;
        v8f c = {};
#pragma unroll
        for (int cs = 0; cs < 4; ++cs) {
          v16h bf;
          if (UW) {
            bf = *(const v16h*)(wsW1 + (((size_t)ntile * 4 + cs) * 32 + lane) * 16);
          } else {
#pragma unroll
            for (int i = 0; i < 16; ++i) {
              int k = 32 * cs + (hi ? 16 + i : i);
              bf[i] = (_Float16) w1sW[(size_t)k * 4096 + ncol];
            }
          }
          c = wmma16(aF[cs], bf, c);
        }
        const float wu = w1sW[(size_t)128 * 4096 + ncol];   // 129th row
        const float bn = w1sB[ncol];
#pragma unroll
        for (int j = 0; j < 8; ++j) {
          int m = j + (hi ? 8 : 0);
          float v = c[j] + sUnc[m] * wu + bn;
          sCst[(w * 16 + m) * GAT_D + t * 16 + l16] = (_Float16) v;
        }
      }
      asm volatile("s_wait_dscnt 0" ::: "memory");
      {
        const int brow = l16;
        const int nh = hi ? 1 : 0;
        float crow[32];
#pragma unroll
        for (int d = 0; d < 32; ++d)
          crow[d] = (float) sCst[(w * 16 + brow) * GAT_D + d];
#pragma unroll
        for (int j = 0; j < 8; ++j) {
          const int n = nh * 8 + j;
          float dot = 0.f;
#pragma unroll
          for (int d = 0; d < 32; ++d)
            dot += crow[d] * (float) sG[(brow * 16 + n) * GAT_D + d];
          w1acc[ei * 8 + j] += 0.25f * fabsf(dot);
        }
      }
    }
  }
  __syncthreads();

  // hidden[b,e] = elu( sum_n q[b,n]*w1[b,n,e] + b1[b,e] )
  {
    const int brow = l16;
    const int nh = hi ? 1 : 0;
#pragma unroll
    for (int ei = 0; ei < 4; ++ei) {
      float p = 0.f;
#pragma unroll
      for (int j = 0; j < 8; ++j)
        p += sQ[brow * 16 + nh * 8 + j] * w1acc[ei * 8 + j];
      sPart[((w * 2 + nh) * 16 + brow) * 4 + ei] = p;
    }
    asm volatile("s_wait_dscnt 0" ::: "memory");
    if (!hi) {
#pragma unroll
      for (int ei = 0; ei < 4; ++ei) {
        const int eidx = w * 4 + ei;
        float tot = sPart[((w * 2 + 0) * 16 + brow) * 4 + ei]
                  + sPart[((w * 2 + 1) * 16 + brow) * 4 + ei];
        sHid[brow * EMB + eidx] = elu1(tot + sB1[brow * EMB + eidx]);
      }
    }
  }
  __syncthreads();

  if (tid < 16) {
    float acc = sV[tid];
    for (int e = 0; e < EMB; ++e) acc += sHid[tid * EMB + e] * sWf[tid * EMB + e];
    out[row0 + tid] = acc;
  }
}

// ---------------------------------------------------------------------------
extern "C" void kernel_launch(void* const* d_in, const int* in_sizes, int n_in,
                              void* d_out, int out_size, void* d_ws, size_t ws_size,
                              hipStream_t stream) {
  (void)n_in; (void)out_size;
  const float* qs  = (const float*)d_in[0];
  const float* st  = (const float*)d_in[1];
  const float* hs  = (const float*)d_in[2];
  const float* un  = (const float*)d_in[3];
  const float* Wg  = (const float*)d_in[4];
  const float* aA  = (const float*)d_in[5];
  const float* w1W = (const float*)d_in[6];
  const float* w1b = (const float*)d_in[7];
  const float* b1W = (const float*)d_in[8];
  const float* b1b = (const float*)d_in[9];
  const float* wfW = (const float*)d_in[10];
  const float* wfb = (const float*)d_in[11];
  const float* V1W = (const float*)d_in[12];
  const float* V1b = (const float*)d_in[13];
  const float* V2W = (const float*)d_in[14];
  const float* V2b = (const float*)d_in[15];
  float* out = (float*)d_out;

  const int Brows = in_sizes[3];           // 4096 rows
  const int grid  = Brows / ROWS_PER_BLK;  // 256 blocks

  // Workspace layout (all f16 fragments)
  const size_t nHS = (size_t)Brows * 1024;       // 4,194,304 halfs
  const size_t nW1 = 524288;
  const size_t nWg = 8192;
  const size_t nSm = 12288;
  const size_t need = (nHS + nW1 + nWg + nSm) * sizeof(_Float16);

  if (d_ws && ws_size >= need) {
    _Float16* wsHS = (_Float16*)d_ws;
    _Float16* wsW1 = wsHS + nHS;
    _Float16* wsWg = wsW1 + nW1;
    _Float16* wsSm = wsWg + nWg;

    swz_hs  <<<dim3((unsigned)(nHS / 256)), dim3(256), 0, stream>>>(hs, wsHS);
    swz_w1s <<<dim3((unsigned)(nW1 / 256)), dim3(256), 0, stream>>>(w1W, wsW1);
    swz_small<<<dim3((unsigned)((nWg + nSm) / 256)), dim3(256), 0, stream>>>(
        Wg, b1W, wfW, V1W, wsWg, wsSm);

    dvd_mixer_fused<true><<<dim3(grid), dim3(NTHREADS), 0, stream>>>(
        qs, st, hs, un, Wg, aA, w1W, w1b, b1W, b1b, wfW, wfb, V1W, V1b, V2W, V2b,
        wsHS, wsW1, wsWg, wsSm, out);
  } else {
    dvd_mixer_fused<false><<<dim3(grid), dim3(NTHREADS), 0, stream>>>(
        qs, st, hs, un, Wg, aA, w1W, w1b, b1W, b1b, wfW, wfb, V1W, V1b, V2W, V2b,
        nullptr, nullptr, nullptr, nullptr, out);
  }
}
